// PatchEmbedLGP_14568529068687
// MI455X (gfx1250) — compile-verified, matrix operand order checked by reading, not run on hardware
//
#include <hip/hip_runtime.h>

typedef __attribute__((ext_vector_type(16))) __bf16 v16bf;
typedef __attribute__((ext_vector_type(8)))  float  v8f;
typedef int v4i __attribute__((vector_size(16)));

#define TOK   12544      // 64 * 14 * 14 tokens
#define DCH   768
#define KDIM  768

#if __has_builtin(__builtin_amdgcn_global_load_async_to_lds_b128) && \
    __has_builtin(__builtin_amdgcn_s_wait_asynccnt)
#define USE_ASYNC_LDS 1
#else
#define USE_ASYNC_LDS 0
#endif

__device__ __forceinline__ float gelu_f(float x) {
  return 0.5f * x * (1.0f + erff(x * 0.70710678118654752f));
}

// 16B global -> LDS copy: async (ASYNCcnt-tracked) when available, sync otherwise.
__device__ __forceinline__ void cp16_g2l(const __bf16* g, __bf16* l) {
#if USE_ASYNC_LDS
  __builtin_amdgcn_global_load_async_to_lds_b128(
      (__attribute__((address_space(1))) v4i*)const_cast<__bf16*>(g),
      (__attribute__((address_space(3))) v4i*)l, 0, 0);
#else
  *(uint4*)l = *(const uint4*)g;
#endif
}

// ---------------- weight prep ----------------
__global__ void f32_to_bf16(const float* __restrict__ in, __bf16* __restrict__ out, int n) {
  int i = blockIdx.x * blockDim.x + threadIdx.x;
  if (i < n) out[i] = (__bf16)in[i];
}

// Fold 4-level Haar wavelet packet (Walsh block transform on 16x16 patches, scale 1/16)
// into pg_w1: W2[d][c*256 + r*16 + t] = (1/16) * sum_q sign(q,r,t) * pg_w1[d][q*3+c]
__global__ void fold_wavelet_w1(const float* __restrict__ pg_w1, __bf16* __restrict__ W2) {
  int idx = blockIdx.x * blockDim.x + threadIdx.x;
  if (idx >= DCH * KDIM) return;
  int d = idx / KDIM;
  int k = idx % KDIM;
  int c = k >> 8;
  int r = (k >> 4) & 15;
  int t = k & 15;
  float s = 0.0f;
  for (int q = 0; q < 256; ++q) {
    int s1 = (q >> 6) & 3, s2 = (q >> 4) & 3, s3 = (q >> 2) & 3, s4 = q & 3;
    int wm = (s1 >> 1) | ((s2 >> 1) << 1) | ((s3 >> 1) << 2) | ((s4 >> 1) << 3);
    int hm = (s1 & 1) | ((s2 & 1) << 1) | ((s3 & 1) << 2) | ((s4 & 1) << 3);
    int sgn = __popc(wm & t) + __popc(hm & r);
    float v = pg_w1[d * 768 + q * 3 + c];
    s += (sgn & 1) ? -v : v;
  }
  W2[idx] = (__bf16)(s * 0.0625f);
}

// im2col of 16x16 stride-16 patches, k = c*256 + r*16 + t  (matches pl_w flat layout)
__global__ void im2col_bf16(const float* __restrict__ x, __bf16* __restrict__ Xp) {
  int idx = blockIdx.x * blockDim.x + threadIdx.x;
  if (idx >= TOK * KDIM) return;
  int tok = idx / KDIM, k = idx % KDIM;
  int b = tok / 196, p = tok % 196, i = p / 14, j = p % 14;
  int c = k >> 8, r = (k >> 4) & 15, t = k & 15;
  Xp[idx] = (__bf16)x[((size_t)(b * 3 + c) * 224 + (size_t)(i * 16 + r)) * 224 + (j * 16 + t)];
}

// ---------------- WMMA GEMM: C(M x 768) = A(M x 768) * W(768 x 768)^T + epilogue ----
// Block = 128 threads (4 waves). Block tile 128(M) x 64(N); each wave computes 32x64
// (2 A-frags x 4 B-frags = 8 WMMAs per K-step). Double-buffered LDS with async copies.
enum { EP_GELU_F32 = 0, EP_F32 = 1, EP_GELU_BF16 = 2, EP_MIX = 3, EP_RELU_F32 = 4, EP_RELU_ADD = 5 };

template <int EP>
__global__ __launch_bounds__(128)
void gemm_wmma(const __bf16* __restrict__ A,
               const __bf16* __restrict__ W,      // row n holds weights for output channel n
               const float* __restrict__ bias,
               const float* __restrict__ extra,   // loc (for MIX / RELU_ADD)
               const float* __restrict__ alpha,   // per-channel gate (MIX)
               float* __restrict__ outF,
               __bf16* __restrict__ outB)
{
  constexpr int TK = 32, TM = 128, TN = 64, LDSTR = TK + 8;  // 80B row stride (padded)
  constexpr int NKT = KDIM / TK;                             // 24 K-tiles
  __shared__ alignas(16) __bf16 sA[2][TM * LDSTR];
  __shared__ alignas(16) __bf16 sW[2][TN * LDSTR];
  const int tid    = threadIdx.x;
  const int wave   = tid >> 5, lane = tid & 31;
  const int lane15 = lane & 15, grp = lane >> 4;
  const int m0 = blockIdx.x * TM, n0 = blockIdx.y * TN;

  // 6 x 16B copies per thread per K-tile: A = 128x32 halves (512 chunks), W = 64x32 (256)
  auto load_tile = [&](int buf, int k0) {
    #pragma unroll
    for (int it = 0; it < 4; ++it) {
      int c = tid + it * 128;
      int row = c >> 2, colh = (c & 3) * 8;
      cp16_g2l(&A[(size_t)(m0 + row) * KDIM + k0 + colh], &sA[buf][row * LDSTR + colh]);
    }
    #pragma unroll
    for (int it = 0; it < 2; ++it) {
      int c = tid + it * 128;
      int row = c >> 2, colh = (c & 3) * 8;
      cp16_g2l(&W[(size_t)(n0 + row) * KDIM + k0 + colh], &sW[buf][row * LDSTR + colh]);
    }
#if !USE_ASYNC_LDS
    if (k0 + TK < KDIM) {
      __builtin_prefetch(&A[(size_t)(m0 + (tid >> 2)) * KDIM + k0 + TK], 0, 1);
      __builtin_prefetch(&W[(size_t)(n0 + (tid >> 2)) * KDIM + k0 + TK], 0, 1);
    }
#endif
  };

  v8f acc[2][4];
  #pragma unroll
  for (int mi = 0; mi < 2; ++mi)
    #pragma unroll
    for (int s = 0; s < 4; ++s)
      #pragma unroll
      for (int i = 0; i < 8; ++i) acc[mi][s][i] = 0.0f;

  load_tile(0, 0);

  for (int kt = 0; kt < NKT; ++kt) {
    const int cur = kt & 1;
    __syncthreads();                       // buf[cur^1] safe to overwrite (dscnt flushed)
#if USE_ASYNC_LDS
    if (kt + 1 < NKT) {
      load_tile(cur ^ 1, (kt + 1) * TK);
      __builtin_amdgcn_s_wait_asynccnt(6); // current tile's 6 async copies complete
    } else {
      __builtin_amdgcn_s_wait_asynccnt(0);
    }
#else
    if (kt + 1 < NKT) load_tile(cur ^ 1, (kt + 1) * TK);
#endif
    __syncthreads();                       // current tile visible to all waves

    // A fragments (16x32 bf16): lane grp 0 -> K[0..7],K[16..23]; grp 1 -> K[8..15],K[24..31]
    v16bf af[2], bf[4];
    #pragma unroll
    for (int mi = 0; mi < 2; ++mi) {
      const __bf16* ar = &sA[cur][(wave * 32 + mi * 16 + lane15) * LDSTR + grp * 8];
      ((uint4*)&af[mi])[0] = *(const uint4*)(ar);
      ((uint4*)&af[mi])[1] = *(const uint4*)(ar + 16);
    }
    #pragma unroll
    for (int s = 0; s < 4; ++s) {
      const __bf16* br = &sW[cur][(s * 16 + lane15) * LDSTR + grp * 8];
      ((uint4*)&bf[s])[0] = *(const uint4*)(br);
      ((uint4*)&bf[s])[1] = *(const uint4*)(br + 16);
    }
    #pragma unroll
    for (int mi = 0; mi < 2; ++mi)
      #pragma unroll
      for (int s = 0; s < 4; ++s)
        acc[mi][s] = __builtin_amdgcn_wmma_f32_16x16x32_bf16(false, af[mi], false, bf[s],
                                                             (short)0, acc[mi][s], false, false);
  }

  // D layout: lanes 0-15: N=lane, M=r; lanes 16-31: N=lane-16, M=8+r
  #pragma unroll
  for (int s = 0; s < 4; ++s) {
    const int n = n0 + s * 16 + lane15;
    const float bv = bias[n];
    float gate = 0.0f;
    if (EP == EP_MIX) gate = 0.5f * (tanhf(alpha[n]) + 1.0f);
    #pragma unroll
    for (int mi = 0; mi < 2; ++mi) {
      const int mbase = m0 + wave * 32 + mi * 16 + grp * 8;
      #pragma unroll
      for (int r = 0; r < 8; ++r) {
        const size_t o = (size_t)(mbase + r) * DCH + n;
        float v = acc[mi][s][r] + bv;
        if (EP == EP_GELU_F32)       outF[o] = gelu_f(v);
        else if (EP == EP_F32)       outF[o] = v;
        else if (EP == EP_GELU_BF16) outB[o] = (__bf16)gelu_f(v);
        else if (EP == EP_MIX)       outF[o] = (1.0f - gate) * extra[o] + gate * v;
        else if (EP == EP_RELU_F32)  outF[o] = fmaxf(v, 0.0f);
        else if (EP == EP_RELU_ADD)  outF[o] = extra[o] + fmaxf(v, 0.0f);
      }
    }
  }
}

// ---------------- depthwise 3x3, pad 1, token-major (T x 768), bf16 out -----------
__global__ void dwconv3x3(const float* __restrict__ in, const float* __restrict__ w,
                          __bf16* __restrict__ out)
{
  int idx = blockIdx.x * blockDim.x + threadIdx.x;
  if (idx >= TOK * (DCH / 4)) return;
  int cg = idx % (DCH / 4);
  int tok = idx / (DCH / 4);
  int b = tok / 196, p = tok % 196, i = p / 14, j = p % 14;
  int d = cg * 4;
  float wv[4][9];
  #pragma unroll
  for (int u = 0; u < 4; ++u)
    #pragma unroll
    for (int q = 0; q < 9; ++q) wv[u][q] = w[(d + u) * 9 + q];
  float a0 = 0, a1 = 0, a2 = 0, a3 = 0;
  #pragma unroll
  for (int dy = 0; dy < 3; ++dy) {
    int ii = i + dy - 1;
    if (ii < 0 || ii >= 14) continue;
    #pragma unroll
    for (int dx = 0; dx < 3; ++dx) {
      int jj = j + dx - 1;
      if (jj < 0 || jj >= 14) continue;
      const float4 v = *(const float4*)&in[(size_t)(b * 196 + ii * 14 + jj) * DCH + d];
      int q = dy * 3 + dx;
      a0 += wv[0][q] * v.x; a1 += wv[1][q] * v.y;
      a2 += wv[2][q] * v.z; a3 += wv[3][q] * v.w;
    }
  }
  size_t o = (size_t)tok * DCH + d;
  out[o] = (__bf16)a0; out[o + 1] = (__bf16)a1;
  out[o + 2] = (__bf16)a2; out[o + 3] = (__bf16)a3;
}

// ---------------- LayerNorm over D per token (one wave32 per token) + write out ---
__global__ void layernorm_write(const float* __restrict__ h, const float* __restrict__ g,
                                const float* __restrict__ be, float* __restrict__ out)
{
  int wid = (int)((blockIdx.x * blockDim.x + threadIdx.x) >> 5);
  int lane = threadIdx.x & 31;
  if (wid >= TOK) return;
  const float* row = h + (size_t)wid * DCH;
  float s = 0.0f, s2 = 0.0f;
  for (int d = lane; d < DCH; d += 32) { float v = row[d]; s += v; s2 += v * v; }
  #pragma unroll
  for (int off = 16; off > 0; off >>= 1) { s += __shfl_xor(s, off); s2 += __shfl_xor(s2, off); }
  float mu = s * (1.0f / DCH);
  float var = s2 * (1.0f / DCH) - mu * mu;
  float inv = rsqrtf(var + 1e-5f);
  int b = wid / 196, p = wid % 196;
  float* o = out + ((size_t)b * 197 + 1 + p) * DCH;
  for (int d = lane; d < DCH; d += 32) o[d] = (row[d] - mu) * inv * g[d] + be[d];
}

__global__ void write_cls(const float* __restrict__ cls, float* __restrict__ out) {
  int idx = blockIdx.x * blockDim.x + threadIdx.x;
  if (idx >= 64 * DCH) return;
  int b = idx / DCH, d = idx % DCH;
  out[(size_t)b * 197 * DCH + d] = cls[d];
}

// ---------------- driver ----------------
extern "C" void kernel_launch(void* const* d_in, const int* in_sizes, int n_in,
                              void* d_out, int out_size, void* d_ws, size_t ws_size,
                              hipStream_t stream)
{
  (void)in_sizes; (void)n_in; (void)out_size; (void)ws_size;
  const float* x      = (const float*)d_in[0];
  const float* pl_w   = (const float*)d_in[1];
  const float* pl_b   = (const float*)d_in[2];
  const float* pl_dw  = (const float*)d_in[3];
  const float* pl_pw  = (const float*)d_in[4];
  const float* pl_pwb = (const float*)d_in[5];
  const float* pg_w1  = (const float*)d_in[6];
  const float* pg_b1  = (const float*)d_in[7];
  const float* pg_dw  = (const float*)d_in[8];
  const float* pg_pw  = (const float*)d_in[9];
  const float* pg_pwb = (const float*)d_in[10];
  const float* pg_w2  = (const float*)d_in[11];
  const float* pg_b2  = (const float*)d_in[12];
  const float* alpha  = (const float*)d_in[13];
  const float* f1_dw  = (const float*)d_in[14];
  const float* f1_pw  = (const float*)d_in[15];
  const float* f1_pwb = (const float*)d_in[16];
  const float* f2_dw  = (const float*)d_in[17];
  const float* f2_pw  = (const float*)d_in[18];
  const float* f2_pwb = (const float*)d_in[19];
  const float* f3_dw  = (const float*)d_in[20];
  const float* f3_pw  = (const float*)d_in[21];
  const float* f3_pwb = (const float*)d_in[22];
  const float* ln_g   = (const float*)d_in[23];
  const float* ln_b   = (const float*)d_in[24];
  const float* cls    = (const float*)d_in[25];
  float* out = (float*)d_out;

  char* ws = (char*)d_ws;
  size_t off = 0;
  auto take = [&](size_t bytes) -> void* {
    void* p = ws + off;
    off += (bytes + 255) & ~(size_t)255;
    return p;
  };
  const size_t WN = (size_t)DCH * KDIM;          // 589824
  __bf16* Xp    = (__bf16*)take((size_t)TOK * KDIM * 2);
  __bf16* W2    = (__bf16*)take(WN * 2);
  __bf16* Wplw  = (__bf16*)take(WN * 2);
  __bf16* Wplpw = (__bf16*)take(WN * 2);
  __bf16* Wpgpw = (__bf16*)take(WN * 2);
  __bf16* Wpgw2 = (__bf16*)take(WN * 2);
  __bf16* Wf1   = (__bf16*)take(WN * 2);
  __bf16* Wf2   = (__bf16*)take(WN * 2);
  __bf16* Wf3   = (__bf16*)take(WN * 2);
  float*  FA    = (float*)take((size_t)TOK * DCH * 4);
  float*  FLOC  = (float*)take((size_t)TOK * DCH * 4);
  __bf16* DB1   = (__bf16*)take((size_t)TOK * DCH * 2);
  __bf16* DB2   = (__bf16*)take((size_t)TOK * DCH * 2);

  const int CB = 256;
  const int wblk = (int)((WN + CB - 1) / CB);
  f32_to_bf16<<<wblk, CB, 0, stream>>>(pl_w,  Wplw,  (int)WN);
  f32_to_bf16<<<wblk, CB, 0, stream>>>(pl_pw, Wplpw, (int)WN);
  f32_to_bf16<<<wblk, CB, 0, stream>>>(pg_pw, Wpgpw, (int)WN);
  f32_to_bf16<<<wblk, CB, 0, stream>>>(pg_w2, Wpgw2, (int)WN);
  f32_to_bf16<<<wblk, CB, 0, stream>>>(f1_pw, Wf1,   (int)WN);
  f32_to_bf16<<<wblk, CB, 0, stream>>>(f2_pw, Wf2,   (int)WN);
  f32_to_bf16<<<wblk, CB, 0, stream>>>(f3_pw, Wf3,   (int)WN);
  fold_wavelet_w1<<<wblk, CB, 0, stream>>>(pg_w1, W2);

  im2col_bf16<<<(TOK * KDIM) / CB, CB, 0, stream>>>(x, Xp);

  dim3 gg(TOK / 128, DCH / 64);   // 98 x 12
  const int dwblk = (TOK * (DCH / 4)) / CB;

  // local path: loc0 -> gelu (fused) -> dw -> pw = loc
  gemm_wmma<EP_GELU_F32><<<gg, 128, 0, stream>>>(Xp,  Wplw,  pl_b,   nullptr, nullptr, FA,   nullptr);
  dwconv3x3<<<dwblk, CB, 0, stream>>>(FA, pl_dw, DB1);
  gemm_wmma<EP_F32><<<gg, 128, 0, stream>>>(DB1, Wplpw, pl_pwb, nullptr, nullptr, FLOC, nullptr);

  // global path: (wavelet folded into W2) -> gelu -> dw -> pw -> gelu -> pw + gate mix
  gemm_wmma<EP_GELU_F32><<<gg, 128, 0, stream>>>(Xp,  W2,    pg_b1,  nullptr, nullptr, FA,   nullptr);
  dwconv3x3<<<dwblk, CB, 0, stream>>>(FA, pg_dw, DB1);
  gemm_wmma<EP_GELU_BF16><<<gg, 128, 0, stream>>>(DB1, Wpgpw, pg_pwb, nullptr, nullptr, nullptr, DB2);
  gemm_wmma<EP_MIX><<<gg, 128, 0, stream>>>(DB2, Wpgw2, pg_b2,  FLOC,    alpha,   FA,   nullptr);

  // fuse stack: f1, f2 (with +loc), f3
  dwconv3x3<<<dwblk, CB, 0, stream>>>(FA, f1_dw, DB1);
  gemm_wmma<EP_RELU_F32><<<gg, 128, 0, stream>>>(DB1, Wf1, f1_pwb, nullptr, nullptr, FA, nullptr);
  dwconv3x3<<<dwblk, CB, 0, stream>>>(FA, f2_dw, DB1);
  gemm_wmma<EP_RELU_ADD><<<gg, 128, 0, stream>>>(DB1, Wf2, f2_pwb, FLOC,    nullptr, FA, nullptr);
  dwconv3x3<<<dwblk, CB, 0, stream>>>(FA, f3_dw, DB1);
  gemm_wmma<EP_F32><<<gg, 128, 0, stream>>>(DB1, Wf3, f3_pwb, nullptr, nullptr, FA, nullptr);

  // LayerNorm + output assembly
  layernorm_write<<<(TOK * 32) / CB, CB, 0, stream>>>(FA, ln_g, ln_b, out);
  write_cls<<<(64 * DCH) / CB, CB, 0, stream>>>(cls, out);
}
